// InteracitveNeck_996432412703
// MI455X (gfx1250) — matrix operand order repaired and checked by reading, not compiled, not run on hardware
//
#include <hip/hip_runtime.h>
#include <hip/hip_bf16.h>
#include <stdint.h>
#include <stddef.h>

// ---------------------------------------------------------------------------
// Types for CDNA5 WMMA (wave32): D(16x16 f32) = A(16x32 bf16) x B(32x16 bf16) + C
// ---------------------------------------------------------------------------
typedef __attribute__((ext_vector_type(16))) __bf16    v16bf;
typedef __attribute__((ext_vector_type(8)))  float     v8f;
typedef __attribute__((ext_vector_type(4)))  uint32_t  v4u;
typedef __attribute__((ext_vector_type(8)))  uint32_t  v8u;

__device__ __forceinline__ unsigned short f32_bf16_rne(float f) {
  union { float f; unsigned int u; } cv; cv.f = f;
  unsigned int u = cv.u;
  u += 0x7fffu + ((u >> 16) & 1u);   // round-to-nearest-even
  return (unsigned short)(u >> 16);
}

// ---------------------------------------------------------------------------
// f32 -> bf16 (packed as ushort) flat converter (weights)
// ---------------------------------------------------------------------------
__global__ void cvt_f32_bf16_kernel(const float* __restrict__ src,
                                    unsigned short* __restrict__ dst, int n) {
  int i = blockIdx.x * blockDim.x + threadIdx.x;
  if (i < n) dst[i] = f32_bf16_rne(src[i]);
}

__global__ void zero_kernel(float* __restrict__ p, int n) {
  int i = blockIdx.x * blockDim.x + threadIdx.x;
  if (i < n) p[i] = 0.0f;
}

// ---------------------------------------------------------------------------
// Generic implicit-GEMM convolution via WMMA bf16.
//   input : (NB, Ci, IX, IY, IZ) f32
//   weight: (Co, Ci*KD*KH*KW)   bf16 (pre-converted, natural OIDHW order)
//   out   : (NB, Co, OX, OY, OZ) f32, bias added
// Block tile: M=32 (Co), N=64 (voxels), K-step 64. 128 threads = 4 waves,
// each wave owns one 16-wide N slice and both 16-row M halves -> 2 accs,
// 4 v_wmma per K-step.
// A tile (32 rows x 128B, row pitch K*2 bytes) is moved by the Tensor Data
// Mover: wave 0 builds a D# descriptor in SGPRs and issues tensor_load_to_lds
// (TENSORcnt), overlapped with the VALU im2col gather of the B tile.
// Bias is async-staged to LDS via global_load_async_to_lds_b32 (ASYNCcnt).
// For every conv in this net: Co % 32 == 0 and K % 64 == 0 (no tails).
// ---------------------------------------------------------------------------
__global__ __launch_bounds__(128)
void conv_wmma_kernel(const float* __restrict__ in,
                      const unsigned short* __restrict__ wgt,
                      const float* __restrict__ bias,
                      float* __restrict__ out,
                      int Ci, int Co,
                      int IX, int IY, int IZ,
                      int OX, int OY, int OZ,
                      int KD, int KH, int KW,
                      int sx, int sy, int sz,
                      int px, int py, int pz)
{
  const int K    = Ci * KD * KH * KW;
  const int Ntot = OX * OY * OZ;
  const int n0   = blockIdx.x * 64;
  const int m0   = blockIdx.y * 32;

  in  += (size_t)blockIdx.z * (size_t)Ci * IX * IY * IZ;
  out += (size_t)blockIdx.z * (size_t)Co * Ntot;

  __shared__ unsigned short Asub[32 * 64];   // 32 Co rows x 64 K   (4 KB)
  __shared__ unsigned short Bsub[64 * 64];   // 64 K rows  x 64 vox (8 KB)
  __shared__ int   s_nx[64], s_ny[64], s_nz[64];
  __shared__ float s_bias[32];

  const int tid  = threadIdx.x;   // 0..127
  const int lane = tid & 31;
  const int wave = tid >> 5;

  const int KHW  = KH * KW;
  const int KDHW = KD * KH * KW;
  const int IYZ  = IY * IZ;

  // ---- per-block: decompose the 64 output positions once ----
  if (tid < 64) {
    int n = n0 + tid;
    if (n < Ntot) {
      int oz = n % OZ; int t2 = n / OZ; int oy = t2 % OY; int ox = t2 / OY;
      s_nx[tid] = ox * sx - px;
      s_ny[tid] = oy * sy - py;
      s_nz[tid] = oz * sz - pz;
    } else {
      s_nx[tid] = -0x40000000; s_ny[tid] = -0x40000000; s_nz[tid] = -0x40000000;
    }
  }

  // ---- prologue: async-stage the 32 bias values into LDS (ASYNCcnt) ----
  if (tid < 32) {
    unsigned int loff = (unsigned int)(uintptr_t)&s_bias[tid];
    const float* g = bias + m0 + tid;
    asm volatile("global_load_async_to_lds_b32 %0, %1, off"
                 :: "v"(loff), "v"(g) : "memory");
  }
  if (wave == 0)
    asm volatile("s_wait_asynccnt 0" ::: "memory");

  v8f acc0 = {};
  v8f acc1 = {};

  const int mrow = lane & 15;
  const int half = lane >> 4;
  const int ncol = wave * 16 + (lane & 15);
  const unsigned int asub_lds = (unsigned int)(uintptr_t)&Asub[0];

  for (int k0 = 0; k0 < K; k0 += 64) {
    __syncthreads();   // LDS reuse fence (also publishes s_n*/s_bias on first pass)

    // ---- A tile via Tensor Data Mover: 2D tile 32 dwords x 32 rows ----
    if (wave == 0) {
      uint64_t ga = (uint64_t)(uintptr_t)(wgt + (size_t)m0 * K + k0);
      v4u g0; v8u g1;
      g0[0] = 1u;                                   // count=1 (valid descriptor)
      g0[1] = asub_lds;                             // lds_addr (bytes)
      g0[2] = (uint32_t)ga;                         // global_addr[31:0]
      g0[3] = (uint32_t)((ga >> 32) & 0x01FFFFFFu)  // global_addr[56:32]
              | (2u << 30);                         // type = 2 ("image")
      g1[0] = 0x00020000u;       // wg_mask=0 (not in cluster), data_size=2 (4B)
      g1[1] = 32u << 16;         // tensor_dim0[15:0]=32 dwords
      g1[2] = 32u << 16;         // tensor_dim0 hi=0 | tensor_dim1[15:0]=32
      g1[3] = 32u << 16;         // tensor_dim1 hi=0 | tile_dim0=32 dwords (128B row)
      g1[4] = 32u;               // tile_dim1=32 rows, tile_dim2=0 (2D)
      g1[5] = (uint32_t)(K >> 1);// tensor_dim0_stride[31:0] = K/2 dwords (row pitch)
      g1[6] = 0u;                // stride hi=0 | tensor_dim1_stride lo=0
      g1[7] = 0u;
      asm volatile("tensor_load_to_lds %0, %1" :: "s"(g0), "s"(g1) : "memory");
    }

    // ---- B tile: im2col gather + f32->bf16, overlapped with the TDM DMA ----
    {
      int kkA = tid >> 2;              // 0..31 (and +32)
      int nnB = (tid & 3) * 16;        // 0,16,32,48
      #pragma unroll
      for (int kh2 = 0; kh2 < 2; ++kh2) {
        int kk = kkA + kh2 * 32;
        int k  = k0 + kk;
        int ci = k / KDHW;
        int r  = k - ci * KDHW;
        int kd = r / KHW;  r -= kd * KHW;
        int kh = r / KW;
        int kw = r - kh * KW;
        size_t kbase = (size_t)ci * IX * IYZ;
        #pragma unroll
        for (int j = 0; j < 16; ++j) {
          int nn = nnB + j;
          int ix = s_nx[nn] + kd;
          int iy = s_ny[nn] + kh;
          int iz = s_nz[nn] + kw;
          unsigned short v = 0;
          if ((unsigned)ix < (unsigned)IX && (unsigned)iy < (unsigned)IY &&
              (unsigned)iz < (unsigned)IZ)
            v = f32_bf16_rne(in[kbase + (size_t)ix * IYZ + iy * IZ + iz]);
          Bsub[kk * 64 + nn] = v;
        }
      }
    }

    if (wave == 0)
      __builtin_amdgcn_s_wait_tensorcnt(0);   // TDM tile landed in LDS
    __syncthreads();

    // ---- fragments per ISA 7.12.2 layouts; 4 WMMA per K-step ----
    #pragma unroll
    for (int ks = 0; ks < 2; ++ks) {
      union { v16bf v; unsigned short u[16]; } a0, a1, bfr;
      #pragma unroll
      for (int j = 0; j < 8; ++j) {
        // A 16-bit 16x32: VGPR j holds K = (j<4 ? 2j : 8+2j) + half*8
        int kb = (j < 4 ? 2 * j : 8 + 2 * j) + half * 8 + ks * 32;
        a0.u[2 * j]     = Asub[mrow * 64 + kb];
        a0.u[2 * j + 1] = Asub[mrow * 64 + kb + 1];
        a1.u[2 * j]     = Asub[(16 + mrow) * 64 + kb];
        a1.u[2 * j + 1] = Asub[(16 + mrow) * 64 + kb + 1];
      }
      #pragma unroll
      for (int e = 0; e < 16; ++e) {
        // B 32x16: element e holds K = (e&7) + 8*half + 16*(e>>3)
        bfr.u[e] = Bsub[(ks * 32 + half * 8 + (e & 7) + 16 * (e >> 3)) * 64 + ncol];
      }
      acc0 = __builtin_amdgcn_wmma_f32_16x16x32_bf16(
          false, a0.v, false, bfr.v, (short)0, acc0, false, false);
      acc1 = __builtin_amdgcn_wmma_f32_16x16x32_bf16(
          false, a1.v, false, bfr.v, (short)0, acc1, false, false);
    }
  }

  // C/D 16x16 f32: lane L -> N = L&15 (+wave slice), VGPR i -> M = i + 8*(L>>4)
  int n = n0 + wave * 16 + (lane & 15);
  if (n < Ntot) {
    #pragma unroll
    for (int i = 0; i < 8; ++i) {
      int mi = i + 8 * half;
      int m  = m0 + mi;
      out[(size_t)m * Ntot + n]        = acc0[i] + s_bias[mi];
      out[(size_t)(m + 16) * Ntot + n] = acc1[i] + s_bias[mi + 16];
    }
  }
}

// ---------------------------------------------------------------------------
// BatchNorm (training-style batch stats over all non-channel axes)
// stats[2c]=sum, stats[2c+1]=sumsq ; layout (NB, C, S)
// ---------------------------------------------------------------------------
__global__ void bn_stats_kernel(const float* __restrict__ x, float* __restrict__ stats,
                                int NB, int C, int S) {
  int c = blockIdx.y;
  long tot = (long)NB * S;
  float s = 0.f, s2 = 0.f;
  for (long i = (long)blockIdx.x * blockDim.x + threadIdx.x; i < tot;
       i += (long)gridDim.x * blockDim.x) {
    int n  = (int)(i / S);
    int sp = (int)(i - (long)n * S);
    float v = x[((long)n * C + c) * S + sp];
    s += v; s2 += v * v;
  }
  __shared__ float rs[256], rq[256];
  rs[threadIdx.x] = s; rq[threadIdx.x] = s2;
  __syncthreads();
  for (int off = blockDim.x >> 1; off > 0; off >>= 1) {
    if ((int)threadIdx.x < off) {
      rs[threadIdx.x] += rs[threadIdx.x + off];
      rq[threadIdx.x] += rq[threadIdx.x + off];
    }
    __syncthreads();
  }
  if (threadIdx.x == 0) {
    atomicAdd(&stats[2 * c],     rs[0]);
    atomicAdd(&stats[2 * c + 1], rq[0]);
  }
}

__global__ void bn_apply_kernel(const float* __restrict__ x, const float* __restrict__ stats,
                                const float* __restrict__ g, const float* __restrict__ b,
                                const float* __restrict__ res, float* __restrict__ y,
                                int NB, int C, int S, int relu) {
  long i = (long)blockIdx.x * blockDim.x + threadIdx.x;
  long tot = (long)NB * C * S;
  if (i >= tot) return;
  int c = (int)((i / S) % C);
  float count = (float)NB * (float)S;
  float m   = stats[2 * c] / count;
  float var = stats[2 * c + 1] / count - m * m;
  float v = (x[i] - m) * rsqrtf(var + 1e-5f) * g[c] + b[c];
  if (res)  v += res[i];
  if (relu) v = fmaxf(v, 0.f);
  y[i] = v;
}

// ---------------------------------------------------------------------------
// Camera projection helper (builds proj = K' @ extr[:3,:] per camera)
// ---------------------------------------------------------------------------
__device__ __forceinline__ void project_point(const float* __restrict__ intr,
                                              const float* __restrict__ extr,
                                              int cam, float ratio,
                                              float px, float py, float pz,
                                              int W, int H,
                                              float& ix, float& iy, int& valid) {
  const float* E = extr + cam * 16;
  float P[3][4];
  #pragma unroll
  for (int r = 0; r < 3; ++r) {
    float k0 = intr[r * 3 + 0], k1 = intr[r * 3 + 1], k2 = intr[r * 3 + 2];
    if (r < 2) { k0 /= ratio; k1 /= ratio; k2 /= ratio; }
    #pragma unroll
    for (int j = 0; j < 4; ++j)
      P[r][j] = k0 * E[0 * 4 + j] + k1 * E[1 * 4 + j] + k2 * E[2 * 4 + j];
  }
  float u = P[0][0] * px + P[0][1] * py + P[0][2] * pz + P[0][3];
  float v = P[1][0] * px + P[1][1] * py + P[1][2] * pz + P[1][3];
  float w = P[2][0] * px + P[2][1] * py + P[2][2] * pz + P[2][3];
  float uu = u / (w + 1e-10f), vv = v / (w + 1e-10f);
  valid = (uu >= 0.f) && (vv >= 0.f) && (uu < (float)W) && (vv < (float)H) && (w > 0.f);
  ix = uu * (float)(W - 1) / (float)W;
  iy = vv * (float)(H - 1) / (float)H;
}

__device__ __forceinline__ float bilinear(const float* __restrict__ f, int H, int W,
                                          float ix, float iy) {
  float x0f = floorf(ix), y0f = floorf(iy);
  int x0 = (int)x0f, y0 = (int)y0f;
  float fx = ix - x0f, fy = iy - y0f;
  float v = 0.f;
  #pragma unroll
  for (int dy = 0; dy < 2; ++dy)
    #pragma unroll
    for (int dx = 0; dx < 2; ++dx) {
      int xi = x0 + dx, yi = y0 + dy;
      if ((unsigned)xi < (unsigned)W && (unsigned)yi < (unsigned)H)
        v += (dx ? fx : 1.f - fx) * (dy ? fy : 1.f - fy) * f[yi * W + xi];
    }
  return v;
}

// bilinear sample of the (di,dj)-shifted, zero-padded image (unfold3x3 tap)
__device__ __forceinline__ float bilinear_shift(const float* __restrict__ f, int H, int W,
                                                float ix, float iy, int di, int dj) {
  float x0f = floorf(ix), y0f = floorf(iy);
  int x0 = (int)x0f, y0 = (int)y0f;
  float fx = ix - x0f, fy = iy - y0f;
  float v = 0.f;
  #pragma unroll
  for (int dy = 0; dy < 2; ++dy)
    #pragma unroll
    for (int dx = 0; dx < 2; ++dx) {
      int kx = x0 + dx, ky = y0 + dy;                   // unfold-image coords
      if ((unsigned)kx < (unsigned)W && (unsigned)ky < (unsigned)H) {
        int sxp = kx + dj - 1, syp = ky + di - 1;       // source xf coords
        if ((unsigned)sxp < (unsigned)W && (unsigned)syp < (unsigned)H)
          v += (dx ? fx : 1.f - fx) * (dy ? fy : 1.f - fy) * f[syp * W + sxp];
      }
    }
  return v;
}

// ---------------------------------------------------------------------------
// Initial fov2voxel: average backprojection of (6, C, H, W) -> (C, X, Y, Z)
// blockIdx.x = voxel, threadIdx.x = channel
// ---------------------------------------------------------------------------
__global__ void backproject_avg_kernel(const float* __restrict__ xf,
                                       const float* __restrict__ intr,
                                       const float* __restrict__ extr,
                                       const float* __restrict__ org,
                                       float* __restrict__ out,
                                       int C, int H, int W, int X, int Y, int Z,
                                       float vsx, float vsy, float vsz, float ratio) {
  int vox = blockIdx.x;
  int z = vox % Z; int t = vox / Z; int y = t % Y; int xg = t / Y;
  int c = threadIdx.x;

  __shared__ float s_ix[6], s_iy[6], s_inv;
  __shared__ int   s_valid[6];

  if (c < 6) {
    float px = xg * vsx + org[0] - (float)X * 0.5f * vsx;
    float py = y  * vsy + org[1] - (float)Y * 0.5f * vsy;
    float pz = z  * vsz + org[2] - (float)Z * 0.5f * vsz;
    project_point(intr, extr, c, ratio, px, py, pz, W, H, s_ix[c], s_iy[c], s_valid[c]);
  }
  __syncthreads();
  if (c == 0) {
    float cnt = 0.f;
    for (int n = 0; n < 6; ++n) cnt += (float)s_valid[n];
    s_inv = (cnt != 0.f) ? 1.f / (cnt + 1e-10f) : 0.f;
  }
  __syncthreads();

  float acc = 0.f;
  for (int n = 0; n < 6; ++n)
    acc += bilinear(xf + ((size_t)n * C + c) * H * W, H, W, s_ix[n], s_iy[n]);
  out[(size_t)c * (X * Y * Z) + vox] = acc * s_inv;
}

// ---------------------------------------------------------------------------
// Fused: depthwise 3x3x3 attention conv + unfold3x3 backprojection + weighted sum
//   voxin : (C, X, Y, Z)  post-block voxel (attn input)
//   out   : (C, X, Y, Z)  sum_t attn[c*9+t] * fov2voxel(unfold(xf))[c*9+t]
//   valids: optional (6, X, Z, Y) float mask output (stage 2 -> d_out tail)
// ---------------------------------------------------------------------------
__global__ void unfold_attn_kernel(const float* __restrict__ xf,
                                   const float* __restrict__ voxin,
                                   const float* __restrict__ aw,
                                   const float* __restrict__ ab,
                                   const float* __restrict__ intr,
                                   const float* __restrict__ extr,
                                   const float* __restrict__ org,
                                   float* __restrict__ out,
                                   float* __restrict__ valids,
                                   int C, int H, int W, int X, int Y, int Z,
                                   float vsx, float vsy, float vsz, float ratio) {
  int vox = blockIdx.x;
  int z = vox % Z; int t = vox / Z; int y = t % Y; int xg = t / Y;
  int c = threadIdx.x;

  __shared__ float s_ix[6], s_iy[6], s_inv;
  __shared__ int   s_valid[6];

  if (c < 6) {
    float px = xg * vsx + org[0] - (float)X * 0.5f * vsx;
    float py = y  * vsy + org[1] - (float)Y * 0.5f * vsy;
    float pz = z  * vsz + org[2] - (float)Z * 0.5f * vsz;
    project_point(intr, extr, c, ratio, px, py, pz, W, H, s_ix[c], s_iy[c], s_valid[c]);
  }
  __syncthreads();
  if (c == 0) {
    float cnt = 0.f;
    for (int n = 0; n < 6; ++n) cnt += (float)s_valid[n];
    s_inv = (cnt != 0.f) ? 1.f / (cnt + 1e-10f) : 0.f;
    if (valids)
      for (int n = 0; n < 6; ++n)
        valids[(((size_t)n * X + xg) * Z + z) * Y + y] = (float)s_valid[n];
  }
  __syncthreads();

  // 9 attention taps for channel c (grouped conv: out ch c*9+t <- in ch c)
  float at[9];
  #pragma unroll
  for (int t9 = 0; t9 < 9; ++t9) at[t9] = ab[c * 9 + t9];
  for (int kd = 0; kd < 3; ++kd)
    for (int kh = 0; kh < 3; ++kh)
      for (int kw = 0; kw < 3; ++kw) {
        int xx = xg - 1 + kd, yy = y - 1 + kh, zz = z - 1 + kw;
        if ((unsigned)xx < (unsigned)X && (unsigned)yy < (unsigned)Y &&
            (unsigned)zz < (unsigned)Z) {
          float v = voxin[(((size_t)c * X + xx) * Y + yy) * Z + zz];
          int widx = (kd * 3 + kh) * 3 + kw;
          #pragma unroll
          for (int t9 = 0; t9 < 9; ++t9)
            at[t9] += aw[((size_t)(c * 9 + t9)) * 27 + widx] * v;
        }
      }

  // attention-weighted sum of the 9 backprojected unfold taps, camera-averaged
  float acc = 0.f;
  for (int n = 0; n < 6; ++n) {
    const float* f = xf + ((size_t)n * C + c) * H * W;
    #pragma unroll
    for (int ti = 0; ti < 3; ++ti)
      #pragma unroll
      for (int tj = 0; tj < 3; ++tj)
        acc += at[ti * 3 + tj] * bilinear_shift(f, H, W, s_ix[n], s_iy[n], ti, tj);
  }
  out[(size_t)c * (X * Y * Z) + vox] = acc * s_inv;
}

// ---------------------------------------------------------------------------
// Final: out[c][y][x] = voxel[c][x][y][0]  (swapaxes of (1,256,64,64,1)[...,0])
// ---------------------------------------------------------------------------
__global__ void final_permute_kernel(const float* __restrict__ v, float* __restrict__ out) {
  __builtin_prefetch(v, 0, 1);            // CDNA5 global_prefetch_b8
  int i = blockIdx.x * blockDim.x + threadIdx.x;
  if (i < 256 * 64 * 64) {
    int xcol = i & 63;
    int yrow = (i >> 6) & 63;
    int c    = i >> 12;
    out[(c << 12) + (yrow << 6) + xcol] = v[(c << 12) + (xcol << 6) + yrow];
  }
  __builtin_amdgcn_s_wait_tensorcnt(0);   // CDNA5 split wait
  __builtin_amdgcn_s_cluster_barrier();   // NOP when ClusterID==0
}

// ---------------------------------------------------------------------------
// Host orchestration
// ---------------------------------------------------------------------------
extern "C" void kernel_launch(void* const* d_in, const int* in_sizes, int n_in,
                              void* d_out, int out_size, void* d_ws, size_t ws_size,
                              hipStream_t stream) {
  (void)in_sizes; (void)n_in; (void)out_size; (void)ws_size;

  int p = 0;
  const float* x      = (const float*)d_in[p++];   // (1,6,64,58,100)
  const float* intr   = (const float*)d_in[p++];   // (3,3)
  const float* extr   = (const float*)d_in[p++];   // (6,4,4)
  const float* origin = (const float*)d_in[p++];   // (3,)
  const float *blk_w[3], *blk_b[3], *blk_g[3], *blk_bb[3];
  for (int i = 0; i < 3; ++i) {
    blk_w[i]  = (const float*)d_in[p++]; blk_b[i]  = (const float*)d_in[p++];
    blk_g[i]  = (const float*)d_in[p++]; blk_bb[i] = (const float*)d_in[p++];
  }
  const float *aw_[3], *ab_[3];
  for (int i = 0; i < 3; ++i) {
    aw_[i] = (const float*)d_in[p++]; ab_[i] = (const float*)d_in[p++];
  }
  const float *n1g[3], *n1b[3];
  for (int i = 0; i < 3; ++i) {
    n1g[i] = (const float*)d_in[p++]; n1b[i] = (const float*)d_in[p++];
  }
  const float *d1w[3], *d1b[3], *d1g[3], *d1bb[3];
  for (int i = 0; i < 3; ++i) {
    d1w[i] = (const float*)d_in[p++]; d1b[i]  = (const float*)d_in[p++];
    d1g[i] = (const float*)d_in[p++]; d1bb[i] = (const float*)d_in[p++];
  }
  const float *d2w[3], *d2b[3], *d2g[3], *d2bb[3];
  for (int i = 0; i < 3; ++i) {
    d2w[i] = (const float*)d_in[p++]; d2b[i]  = (const float*)d_in[p++];
    d2g[i] = (const float*)d_in[p++]; d2bb[i] = (const float*)d_in[p++];
  }
  // d_in[p], d_in[p+1] = img_h, img_w (928, 1600 — baked into derived dims)

  // ---- workspace carve ----
  uint8_t* wsp = (uint8_t*)d_ws;
  size_t cur = 0;
  auto carve = [&](size_t bytes) -> void* {
    void* r = wsp + cur;
    cur += (bytes + 255) & ~(size_t)255;
    return r;
  };
  const size_t VOXE = (size_t)256 * 64 * 64 * 3;  // 3,145,728 f32 (stage max)
  float* vA  = (float*)carve(VOXE * 4);
  float* vB  = (float*)carve(VOXE * 4);
  float* vC  = (float*)carve(VOXE * 4);
  float* xf1 = (float*)carve((size_t)6 * 128 * 29 * 50 * 4);
  float* xf2 = (float*)carve((size_t)6 * 256 * 15 * 25 * 4);
  unsigned short* wbf = (unsigned short*)carve((size_t)256 * 256 * 27 * 2);
  float* stats = (float*)carve((size_t)2 * 512 * 4);

  auto launch_bn = [&](const float* xin, float* yout, const float* res,
                       const float* g, const float* b, int NB, int C, int S, int relu) {
    zero_kernel<<<1, 512, 0, stream>>>(stats, 2 * C);
    bn_stats_kernel<<<dim3(32, C), 256, 0, stream>>>(xin, stats, NB, C, S);
    long tot = (long)NB * C * S;
    bn_apply_kernel<<<(unsigned)((tot + 255) / 256), 256, 0, stream>>>(
        xin, stats, g, b, res, yout, NB, C, S, relu);
  };

  auto launch_conv = [&](const float* in, const float* w_f32, const float* bias,
                         float* outp, int NB, int Ci, int Co,
                         int IX, int IY, int IZ, int OX, int OY, int OZ,
                         int KD, int KH, int KW, int sx, int sy, int sz,
                         int px, int py, int pz) {
    int K  = Ci * KD * KH * KW;
    int wn = Co * K;
    cvt_f32_bf16_kernel<<<(wn + 255) / 256, 256, 0, stream>>>(w_f32, wbf, wn);
    int Ntot = OX * OY * OZ;
    dim3 grid((Ntot + 63) / 64, Co / 32, NB);
    conv_wmma_kernel<<<grid, 128, 0, stream>>>(in, wbf, bias, outp, Ci, Co,
                                               IX, IY, IZ, OX, OY, OZ,
                                               KD, KH, KW, sx, sy, sz, px, py, pz);
  };

  // ---- static stage configuration (img_h=928, img_w=1600) ----
  const int   Cs[3]   = {64, 128, 256};
  const int   Zs[3]   = {12, 6, 3};
  const int   Hs[3]   = {58, 29, 15};
  const int   Ws_[3]  = {100, 50, 25};
  const float szH[3]  = {0.32f, 0.64f, 1.28f};
  const float ratio[3] = {900.f * 16.f / 928.f, 900.f * 32.f / 928.f, 900.f * 64.f / 928.f};
  const int   d1Co[3] = {128, 256, 256};
  const int   d1sz[3] = {2, 2, 1};
  const int   d1pz[3] = {1, 1, 0};
  const int   d1Zo[3] = {6, 3, 1};
  const float* xfp[3] = {x, xf1, xf2};

  // ---- initial fov2voxel: (6,64,58,100) -> (64,64,64,12) ----
  backproject_avg_kernel<<<64 * 64 * 12, 64, 0, stream>>>(
      x, intr, extr, origin, vA, 64, 58, 100, 64, 64, 12,
      0.5f, 0.5f, 0.32f, ratio[0]);

  float* vox = vA; float* tA = vB; float* tB = vC;
  for (int i = 0; i < 3; ++i) {
    const int C = Cs[i], Z = Zs[i];
    const int S = 64 * 64 * Z;

    // block: conv3d(C->C, 3x3x3, s1, p1) -> BN -> +residual -> ReLU
    launch_conv(vox, blk_w[i], blk_b[i], tA, 1, C, C,
                64, 64, Z, 64, 64, Z, 3, 3, 3, 1, 1, 1, 1, 1, 1);
    launch_bn(tA, tB, vox, blk_g[i], blk_bb[i], 1, C, S, 1);  // voxel = tB

    // fused depthwise-attn + unfold3x3 backprojection + weighted sum -> tA
    float* validsPtr = (i == 2) ? ((float*)d_out + 256 * 64 * 64) : nullptr;
    unfold_attn_kernel<<<S, C, 0, stream>>>(
        xfp[i], tB, aw_[i], ab_[i], intr, extr, origin, tA, validsPtr,
        C, Hs[i], Ws_[i], 64, 64, Z, 0.5f, 0.5f, szH[i], ratio[i]);

    // norm1: BN(tA) + residual(tB) + ReLU -> vox
    launch_bn(tA, vox, tB, n1g[i], n1b[i], 1, C, S, 1);       // voxel = vox

    // down1: conv3d(C->d1Co, 3x3x3, stride (1,1,sz), pad (1,1,pz)) -> BN -> ReLU
    launch_conv(vox, d1w[i], d1b[i], tA, 1, C, d1Co[i],
                64, 64, Z, 64, 64, d1Zo[i], 3, 3, 3, 1, 1, d1sz[i], 1, 1, d1pz[i]);
    launch_bn(tA, tB, nullptr, d1g[i], d1bb[i], 1, d1Co[i], 64 * 64 * d1Zo[i], 1);

    { float* nv = tB; tB = vox; vox = nv; }                   // voxel = new

    // down2: conv2d(C->d1Co, 3x3, s2, p1) on (6,C,H,W) -> BN -> ReLU
    // (stage 2's down2 does not affect the outputs — skipped)
    if (i < 2) {
      float* xo = (i == 0) ? xf1 : xf2;
      launch_conv(xfp[i], d2w[i], d2b[i], xo, 6, C, d1Co[i],
                  Hs[i], Ws_[i], 1, Hs[i + 1], Ws_[i + 1], 1,
                  3, 3, 1, 2, 2, 1, 1, 1, 0);
      launch_bn(xo, xo, nullptr, d2g[i], d2bb[i], 6, d1Co[i],
                Hs[i + 1] * Ws_[i + 1], 1);
    }
  }

  // final voxel (256,64,64,1): transpose into d_out
  final_permute_kernel<<<(256 * 64 * 64) / 256, 256, 0, stream>>>(vox, (float*)d_out);
}
